// NestedAdaptationBlock_62354335203527
// MI455X (gfx1250) — compile-verified, matrix operand order checked by reading, not run on hardware
//
#include <hip/hip_runtime.h>
#include <hip/hip_bf16.h>
#include <math.h>

typedef __bf16 bf16;
typedef __attribute__((ext_vector_type(16))) __bf16 v16bf;
typedef __attribute__((ext_vector_type(8)))  float  v8f;
typedef int v4i __attribute__((vector_size(16)));   // matches builtin's int4 param

#define B_   2
#define T_   2048
#define C_   1024
#define H_   16
#define D_   64
#define HID_ 256
#define FF_  4096
#define M_   (B_ * T_)   // 4096 rows

enum { EPI_F32 = 0, EPI_RELU_BF16 = 1, EPI_SIGMOID_F32 = 2, EPI_RELUSQ_BF16 = 3, EPI_ADD_F32 = 4 };

#ifndef __has_builtin
#define __has_builtin(x) 0
#endif

#if __has_builtin(__builtin_amdgcn_global_load_async_to_lds_b128)
#define HAVE_ASYNC_LDS 1
#else
#define HAVE_ASYNC_LDS 0
#endif

#define AS_GLOBAL __attribute__((address_space(1)))
#define AS_LDS    __attribute__((address_space(3)))

// ---- float -> bf16 (round-to-nearest-even) ----
__device__ __forceinline__ bf16 f2bf(float f) {
    unsigned u = __builtin_bit_cast(unsigned, f);
    u += 0x7FFFu + ((u >> 16) & 1u);
    unsigned short s = (unsigned short)(u >> 16);
    return __builtin_bit_cast(bf16, s);
}

__global__ __launch_bounds__(256)
void cvt_f32_bf16_kernel(const float* __restrict__ in, bf16* __restrict__ out, int n) {
    int i = blockIdx.x * blockDim.x + threadIdx.x;
    if (i < n) out[i] = f2bf(in[i]);
}

// 16-byte global -> LDS copy; async (ASYNCcnt) on CDNA5 toolchains that expose it.
__device__ __forceinline__ void copy16_to_lds(const bf16* g, bf16* l) {
#if HAVE_ASYNC_LDS
    __builtin_amdgcn_global_load_async_to_lds_b128(
        (AS_GLOBAL v4i*)g, (AS_LDS v4i*)l, 0, 0);
#else
    *(uint4*)l = *(const uint4*)g;
#endif
}
__device__ __forceinline__ void async_wait_all() {
#if HAVE_ASYNC_LDS
#if __has_builtin(__builtin_amdgcn_s_wait_asynccnt)
    __builtin_amdgcn_s_wait_asynccnt(0);
#else
    asm volatile("s_wait_asynccnt 0x0" ::: "memory");
#endif
#endif
}

__device__ __forceinline__ void epi_store(int mode, float v, size_t idx, float bval,
                                          const float* __restrict__ resid,
                                          float* __restrict__ outF, bf16* __restrict__ outB) {
    if (mode == EPI_F32) {
        outF[idx] = v;
    } else if (mode == EPI_RELU_BF16) {
        float r = v + bval; outB[idx] = f2bf(r > 0.f ? r : 0.f);
    } else if (mode == EPI_SIGMOID_F32) {
        float r = v + bval; outF[idx] = 1.0f / (1.0f + __expf(-r));
    } else if (mode == EPI_RELUSQ_BF16) {
        float r = v > 0.f ? v : 0.f; outB[idx] = f2bf(r * r);
    } else { // EPI_ADD_F32
        outF[idx] = v + resid[idx];
    }
}

// ---------------------------------------------------------------------------
// Register-blocked NT GEMM: out[m,n] = sum_k A[m,k] * W[n,k]
// Block = 256 threads = 8 waves in a 4(M) x 2(N) grid.
// Block tile 256M x 64N; each wave owns 64M x 32N = 4x2 WMMA accumulators.
// K staged in 32-wide chunks through LDS via async global->LDS loads.
// grid = (N/64, M/256)
// ---------------------------------------------------------------------------
#define BM 256
#define BN 64
#define BK 32
#define LROW 48   // bf16 elements per LDS row slot (64B data + 32B pad)

__global__ __launch_bounds__(256)
void gemm_wmma_tiled_kernel(const bf16* __restrict__ A, const bf16* __restrict__ W,
                            const float* __restrict__ bias, const float* __restrict__ resid,
                            float* __restrict__ outF, bf16* __restrict__ outB,
                            int M, int N, int K, int mode)
{
    __shared__ bf16 lA[BM * LROW];   // 24 KB
    __shared__ bf16 lB[BN * LROW];   //  6 KB

    const int tid  = threadIdx.x;
    const int lane = tid & 31;
    const int wave = tid >> 5;
    const int wm   = wave >> 1;   // 0..3 : which 64-row band
    const int wn   = wave & 1;    // 0..1 : which 32-col band
    const int m0   = blockIdx.y * BM;
    const int n0   = blockIdx.x * BN;

    v8f acc[4][2];
#pragma unroll
    for (int s = 0; s < 4; ++s)
#pragma unroll
        for (int t = 0; t < 2; ++t)
            acc[s][t] = (v8f){0.f, 0.f, 0.f, 0.f, 0.f, 0.f, 0.f, 0.f};

    const int lrow = lane & 15;
    const int lko  = (lane >> 4) * 16;   // upper/lower K half within the 32-chunk

    for (int k0 = 0; k0 < K; k0 += BK) {
        // ---- stage A block: 256 rows x 32 bf16 = 1024 x 16B chunks, 4 per thread
#pragma unroll
        for (int i = 0; i < 4; ++i) {
            int idx = tid + i * 256;
            int row = idx >> 2, ch = idx & 3;
            copy16_to_lds(A + (size_t)(m0 + row) * K + k0 + ch * 8,
                          lA + row * LROW + ch * 8);
        }
        // ---- stage B block: 64 rows x 32 bf16 = 256 x 16B chunks, 1 per thread
        {
            int row = tid >> 2, ch = tid & 3;
            copy16_to_lds(W + (size_t)(n0 + row) * K + k0 + ch * 8,
                          lB + row * LROW + ch * 8);
        }
        async_wait_all();
        __syncthreads();

        v16bf aF[4], bF[2];
#pragma unroll
        for (int s = 0; s < 4; ++s)
            aF[s] = *(const v16bf*)(lA + (wm * 64 + s * 16 + lrow) * LROW + lko);
#pragma unroll
        for (int t = 0; t < 2; ++t)
            bF[t] = *(const v16bf*)(lB + (wn * 32 + t * 16 + lrow) * LROW + lko);

#pragma unroll
        for (int s = 0; s < 4; ++s)
#pragma unroll
            for (int t = 0; t < 2; ++t)
                acc[s][t] = __builtin_amdgcn_wmma_f32_16x16x32_bf16(
                    false, aF[s], false, bF[t], (short)0, acc[s][t], false, false);

        __syncthreads();
    }

    // ---- epilogue: C/D layout lane&15 = N, (lane>>4)*8 + j = M
#pragma unroll
    for (int s = 0; s < 4; ++s) {
#pragma unroll
        for (int t = 0; t < 2; ++t) {
            const int n  = n0 + wn * 32 + t * 16 + (lane & 15);
            const int mb = m0 + wm * 64 + s * 16 + (lane >> 4) * 8;
            const float bval = bias ? bias[n] : 0.0f;
#pragma unroll
            for (int j = 0; j < 8; ++j) {
                epi_store(mode, acc[s][t][j], (size_t)(mb + j) * N + n, bval, resid, outF, outB);
            }
        }
    }
}

// ---------------------------------------------------------------------------
// Small-N GEMM (one 16x16 tile per wave), used for the N=16 surprise head.
// grid = (N/16, M/128)
// ---------------------------------------------------------------------------
__global__ __launch_bounds__(256)
void gemm_wmma_small_kernel(const bf16* __restrict__ A, const bf16* __restrict__ W,
                            const float* __restrict__ bias, const float* __restrict__ resid,
                            float* __restrict__ outF, bf16* __restrict__ outB,
                            int M, int N, int K, int mode)
{
    const int lane  = threadIdx.x & 31;
    const int wave  = threadIdx.x >> 5;
    const int tileN = blockIdx.x;
    const int tileM = blockIdx.y * 8 + wave;

    const bf16* ap = A + (size_t)(tileM * 16 + (lane & 15)) * K + (lane >> 4) * 16;
    const bf16* wp = W + (size_t)(tileN * 16 + (lane & 15)) * K + (lane >> 4) * 16;

    v8f acc = {0.f, 0.f, 0.f, 0.f, 0.f, 0.f, 0.f, 0.f};
    for (int k0 = 0; k0 < K; k0 += 32) {
        if (k0 + 256 < K) {
            __builtin_prefetch(ap + k0 + 256, 0, 3);
            __builtin_prefetch(wp + k0 + 256, 0, 3);
        }
        v16bf a = *(const v16bf*)(ap + k0);
        v16bf b = *(const v16bf*)(wp + k0);
        acc = __builtin_amdgcn_wmma_f32_16x16x32_bf16(false, a, false, b, (short)0, acc, false, false);
    }
    const int n  = tileN * 16 + (lane & 15);
    const int mb = tileM * 16 + (lane >> 4) * 8;
    const float bval = bias ? bias[n] : 0.0f;
#pragma unroll
    for (int j = 0; j < 8; ++j)
        epi_store(mode, acc[j], (size_t)(mb + j) * N + n, bval, resid, outF, outB);
}

// ---------------------------------------------------------------------------
// Sequential decay-gated scan:  state = state*dt + gt*(k (x) v);  y = q @ state
// One block per (b,h). 256 threads: c = tid&63 (state column), rg = tid>>6.
// ---------------------------------------------------------------------------
__global__ __launch_bounds__(256)
void scan_kernel(const float* __restrict__ q, const float* __restrict__ k,
                 const float* __restrict__ v, const float* __restrict__ g,
                 bf16* __restrict__ ysb)
{
    const int bh = blockIdx.x;
    const int b  = bh / H_;
    const int h  = bh % H_;
    const int tid = threadIdx.x;
    const int c   = tid & 63;
    const int rg  = tid >> 6;

    __shared__ float sK[D_], sV[D_], sQ[D_];
    __shared__ float yp[4][D_];

    float s[16];
#pragma unroll
    for (int j = 0; j < 16; ++j) s[j] = 0.f;

    for (int t = 0; t < T_; ++t) {
        const size_t row  = (size_t)b * T_ + t;
        const size_t base = row * C_ + (size_t)h * D_;

        if (tid < 64)        sK[tid]       = k[base + tid];
        else if (tid < 128)  sV[tid - 64]  = v[base + tid - 64];
        else if (tid < 192)  sQ[tid - 128] = q[base + tid - 128];
        const float gt = g[row * H_ + h];
        const float dt = 1.0f - gt;
        __syncthreads();

        const float gv = gt * sV[c];
        float p = 0.f;
#pragma unroll
        for (int j = 0; j < 16; ++j) {
            const int r = rg * 16 + j;
            s[j] = s[j] * dt + sK[r] * gv;
            p += sQ[r] * s[j];
        }
        yp[rg][c] = p;
        __syncthreads();

        if (tid < 64) {
            float yv = yp[0][tid] + yp[1][tid] + yp[2][tid] + yp[3][tid];
            ysb[base + tid] = f2bf(yv);
        }
        __syncthreads();
    }
}

// ---------------------------------------------------------------------------
__global__ __launch_bounds__(256)
void rmsnorm_kernel(const float* __restrict__ y, bf16* __restrict__ ynb)
{
    const int row = blockIdx.x;
    const float* yr = y + (size_t)row * C_;
    float ss = 0.f;
    for (int i = threadIdx.x; i < C_; i += 256) { float v = yr[i]; ss += v * v; }
    __shared__ float red[256];
    red[threadIdx.x] = ss;
    __syncthreads();
    for (int st = 128; st > 0; st >>= 1) {
        if (threadIdx.x < st) red[threadIdx.x] += red[threadIdx.x + st];
        __syncthreads();
    }
    const float scale = rsqrtf(red[0] * (1.0f / C_) + 1.1920929e-7f);
    for (int i = threadIdx.x; i < C_; i += 256)
        ynb[(size_t)row * C_ + i] = f2bf(yr[i] * scale);
}

// ---------------------------------------------------------------------------
extern "C" void kernel_launch(void* const* d_in, const int* in_sizes, int n_in,
                              void* d_out, int out_size, void* d_ws, size_t ws_size,
                              hipStream_t stream)
{
    (void)in_sizes; (void)n_in; (void)out_size; (void)ws_size;

    const float* x     = (const float*)d_in[0];
    const float* Wq    = (const float*)d_in[1];
    const float* Wk    = (const float*)d_in[2];
    const float* Wv    = (const float*)d_in[3];
    const float* Wm1   = (const float*)d_in[4];
    const float* bm1   = (const float*)d_in[5];
    const float* Wm2   = (const float*)d_in[6];
    const float* bm2   = (const float*)d_in[7];
    const float* Wproj = (const float*)d_in[8];
    const float* Wfc   = (const float*)d_in[9];
    const float* Wcp   = (const float*)d_in[10];
    float* out = (float*)d_out;

    char* ws = (char*)d_ws;
    size_t off = 0;
    auto alloc = [&](size_t bytes) -> void* {
        void* p = ws + off;
        off += (bytes + 255) & ~(size_t)255;
        return p;
    };
    bf16* xb    = (bf16*)alloc((size_t)M_ * C_ * 2);
    bf16* Wqb   = (bf16*)alloc((size_t)C_ * C_ * 2);
    bf16* Wkb   = (bf16*)alloc((size_t)C_ * C_ * 2);
    bf16* Wvb   = (bf16*)alloc((size_t)C_ * C_ * 2);
    bf16* Wm1b  = (bf16*)alloc((size_t)HID_ * C_ * 2);
    bf16* Wm2b  = (bf16*)alloc((size_t)H_ * HID_ * 2);
    bf16* Wprojb= (bf16*)alloc((size_t)C_ * C_ * 2);
    bf16* Wfcb  = (bf16*)alloc((size_t)FF_ * C_ * 2);
    bf16* Wcpb  = (bf16*)alloc((size_t)C_ * FF_ * 2);
    float* qf   = (float*)alloc((size_t)M_ * C_ * 4);
    float* kf   = (float*)alloc((size_t)M_ * C_ * 4);
    float* vf   = (float*)alloc((size_t)M_ * C_ * 4);
    float* surp = (float*)alloc((size_t)M_ * H_ * 4);
    bf16* h1b   = (bf16*)alloc((size_t)M_ * HID_ * 2);
    bf16* ysb   = (bf16*)alloc((size_t)M_ * C_ * 2);
    float* yf   = (float*)alloc((size_t)M_ * C_ * 4);
    bf16* ynb   = (bf16*)alloc((size_t)M_ * C_ * 2);
    bf16* h2b   = (bf16*)alloc((size_t)M_ * FF_ * 2);

    auto cvt = [&](const float* src, bf16* dst, size_t n) {
        cvt_f32_bf16_kernel<<<dim3((unsigned)((n + 255) / 256)), dim3(256), 0, stream>>>(src, dst, (int)n);
    };
    cvt(x,     xb,     (size_t)M_ * C_);
    cvt(Wq,    Wqb,    (size_t)C_ * C_);
    cvt(Wk,    Wkb,    (size_t)C_ * C_);
    cvt(Wv,    Wvb,    (size_t)C_ * C_);
    cvt(Wm1,   Wm1b,   (size_t)HID_ * C_);
    cvt(Wm2,   Wm2b,   (size_t)H_ * HID_);
    cvt(Wproj, Wprojb, (size_t)C_ * C_);
    cvt(Wfc,   Wfcb,   (size_t)FF_ * C_);
    cvt(Wcp,   Wcpb,   (size_t)C_ * FF_);

    auto gemm = [&](const bf16* A, const bf16* W, const float* bias, const float* resid,
                    float* oF, bf16* oB, int M, int N, int K, int mode) {
        dim3 grid((unsigned)(N / BN), (unsigned)(M / BM));
        gemm_wmma_tiled_kernel<<<grid, dim3(256), 0, stream>>>(A, W, bias, resid, oF, oB, M, N, K, mode);
    };

    // QKV projections
    gemm(xb, Wqb, nullptr, nullptr, qf, nullptr, M_, C_, C_, EPI_F32);
    gemm(xb, Wkb, nullptr, nullptr, kf, nullptr, M_, C_, C_, EPI_F32);
    gemm(xb, Wvb, nullptr, nullptr, vf, nullptr, M_, C_, C_, EPI_F32);

    // meta net: surprise = sigmoid(relu(x@Wm1.T + bm1) @ Wm2.T + bm2)
    gemm(xb, Wm1b, bm1, nullptr, nullptr, h1b, M_, HID_, C_, EPI_RELU_BF16);
    gemm_wmma_small_kernel<<<dim3(1, M_ / 128), dim3(256), 0, stream>>>(
        h1b, Wm2b, bm2, nullptr, surp, nullptr, M_, H_, HID_, EPI_SIGMOID_F32);

    // sequential gated scan over T, parallel over B*H = 32 blocks
    scan_kernel<<<dim3(B_ * H_), dim3(256), 0, stream>>>(qf, kf, vf, surp, ysb);

    // projection (residual base)
    gemm(ysb, Wprojb, nullptr, nullptr, yf, nullptr, M_, C_, C_, EPI_F32);

    // rms norm -> bf16
    rmsnorm_kernel<<<dim3(M_), dim3(256), 0, stream>>>(yf, ynb);

    // squared-relu MLP + residual
    gemm(ynb, Wfcb, nullptr, nullptr, nullptr, h2b, M_, FF_, C_, EPI_RELUSQ_BF16);
    gemm(h2b, Wcpb, nullptr, yf, out, nullptr, M_, C_, FF_, EPI_ADD_F32);
}
